// SiMBA_81243601371747
// MI455X (gfx1250) — compile-verified, matrix operand order checked by reading, not run on hardware
//
#include <hip/hip_runtime.h>
#include <hip/hip_bf16.h>
#include <math.h>

typedef __attribute__((ext_vector_type(16))) __bf16 v16bf;
typedef __attribute__((ext_vector_type(8)))  __bf16 v8bf;
typedef __attribute__((ext_vector_type(8)))  float  v8f;

#define MROWS   4096      // B*L
#define CDIM    192
#define DINNER  384
#define DSTATE  64
#define DTRANK  12
#define LSEQ    1024
#define NBATCH  4
#define LAMBDA_SS 0.01f

// ---------------------------------------------------------------------------
// WMMA fragment load: 16-bit A/B 16x32 layout (CDNA5 ISA 7.12.2)
// lane = 16*half + r ; element j -> K = (j>>3)*16 + half*8 + (j&7)
// => two contiguous 8-element (16B) chunks at k0+half*8 and k0+16+half*8
// ---------------------------------------------------------------------------
__device__ __forceinline__ v16bf load_frag(const __bf16* __restrict__ rowp,
                                           int k0, int hv) {
    const v8bf lo = *(const v8bf*)(rowp + k0 + hv * 8);
    const v8bf hi = *(const v8bf*)(rowp + k0 + 16 + hv * 8);
    v16bf r;
#pragma unroll
    for (int i = 0; i < 8; ++i) { r[i] = lo[i]; r[8 + i] = hi[i]; }
    return r;
}

__device__ __forceinline__ v8f wmma_bf16(v16bf a, v16bf b, v8f c) {
    return __builtin_amdgcn_wmma_f32_16x16x32_bf16(
        false, a, false, b, (short)0, c, false, false);
}

// ---------------------------------------------------------------------------
// Generic bf16 WMMA GEMM:  C[m,n] = sum_k A0[m,k]*W0[n,k]  (+ sign*A1*W1)
// Template: MT = M-tiles of 16 per wave (tile = 16*MT x 64),
//           DUAL = fused complex second product (compile-time: no EXEC
//           divergence around WMMA).
// optional bias[n], activation (0 none, 1 relu, 2 softshrink),
// residual add, fp32 and/or bf16 (zero-padded) outputs.
// Nfull multiple of 64, K multiple of 32, M multiple of 16*MT.
// ---------------------------------------------------------------------------
template <int MT, bool DUAL>
__global__ void __launch_bounds__(128)
gemm_bf16_wmma(const __bf16* __restrict__ A0, int lda,
               const __bf16* __restrict__ W0, int ldw,
               const __bf16* __restrict__ A1,
               const __bf16* __restrict__ W1, float sign,
               const float*  __restrict__ bias,
               const float*  __restrict__ residual, int ldres,
               float*        __restrict__ outF, int ldoF,
               __bf16*       __restrict__ outB, int ldoB,
               int M, int Nfull, int Nvalid, int K, int act) {
    const int lane = threadIdx.x & 31;
    const int wid  = blockIdx.x * (blockDim.x >> 5) + (threadIdx.x >> 5);
    const int ntn  = Nfull >> 6;
    const int mt   = wid / ntn;
    const int nt   = wid % ntn;
    if (mt * 16 * MT >= M) return;
    const int hv = lane >> 4;
    const int r  = lane & 15;

    const __bf16* a0p[MT];
    const __bf16* a1p[MT];
#pragma unroll
    for (int u = 0; u < MT; ++u) {
        const size_t row = (size_t)(mt * 16 * MT + u * 16 + r);
        a0p[u] = A0 + row * (size_t)lda;
        if (DUAL) a1p[u] = A1 + row * (size_t)lda;
    }

    const __bf16* wp0[4];
    const __bf16* wp1[4];
#pragma unroll
    for (int t = 0; t < 4; ++t) {
        const size_t nc = (size_t)(nt * 64 + t * 16 + r);
        wp0[t] = W0 + nc * (size_t)ldw;
        if (DUAL) wp1[t] = W1 + nc * (size_t)ldw;
    }

    v8f acc0[MT][4] = {};
    v8f acc1[MT][4] = {};

    for (int k0 = 0; k0 < K; k0 += 32) {
        v16bf a[MT];
#pragma unroll
        for (int u = 0; u < MT; ++u) a[u] = load_frag(a0p[u], k0, hv);
#pragma unroll
        for (int t = 0; t < 4; ++t) {
            v16bf b = load_frag(wp0[t], k0, hv);
#pragma unroll
            for (int u = 0; u < MT; ++u)
                acc0[u][t] = wmma_bf16(a[u], b, acc0[u][t]);
        }
        if (DUAL) {
            v16bf a1[MT];
#pragma unroll
            for (int u = 0; u < MT; ++u) a1[u] = load_frag(a1p[u], k0, hv);
#pragma unroll
            for (int t = 0; t < 4; ++t) {
                v16bf b1 = load_frag(wp1[t], k0, hv);
#pragma unroll
                for (int u = 0; u < MT; ++u)
                    acc1[u][t] = wmma_bf16(a1[u], b1, acc1[u][t]);
            }
        }
    }

#pragma unroll
    for (int t = 0; t < 4; ++t) {
        const int cc = nt * 64 + t * 16 + r;
        const float bv = (bias && cc < Nvalid) ? bias[cc] : 0.0f;
#pragma unroll
        for (int u = 0; u < MT; ++u) {
#pragma unroll
            for (int i = 0; i < 8; ++i) {
                const int rr = mt * 16 * MT + u * 16 + hv * 8 + i;
                float v = acc0[u][t][i];
                if (DUAL) v += sign * acc1[u][t][i];
                v += bv;
                if (act == 1) {
                    v = fmaxf(v, 0.0f);
                } else if (act == 2) {
                    v = (v > LAMBDA_SS) ? (v - LAMBDA_SS)
                        : ((v < -LAMBDA_SS) ? (v + LAMBDA_SS) : 0.0f);
                }
                if (residual) v += residual[(size_t)rr * ldres + cc];
                if (outF && cc < Nvalid) outF[(size_t)rr * ldoF + cc] = v;
                if (outB) outB[(size_t)rr * ldoB + cc] =
                    (cc < Nvalid) ? (__bf16)v : (__bf16)0.0f;
            }
        }
    }
}

// ---------------------------------------------------------------------------
// weight conversions
// ---------------------------------------------------------------------------
__global__ void conv_f32_to_bf16(const float* __restrict__ src,
                                 __bf16* __restrict__ dst, int n) {
    int i = blockIdx.x * blockDim.x + threadIdx.x;
    if (i < n) dst[i] = (__bf16)src[i];
}

// x_proj_w [140][384] -> padded [192][384] bf16
__global__ void conv_xproj_pad(const float* __restrict__ src,
                               __bf16* __restrict__ dst) {
    int i = blockIdx.x * blockDim.x + threadIdx.x;
    if (i >= 192 * 384) return;
    int n = i / 384, k = i % 384;
    float v = (n < 140) ? src[n * 384 + k] : 0.0f;
    dst[i] = (__bf16)v;
}

// cw[2][4][48][48] (d=K, kout=N) -> wt[comp][b][64(N)][64(K)] bf16, zero-pad
__global__ void conv_cw_transpose(const float* __restrict__ cw1,
                                  const float* __restrict__ cw2,
                                  __bf16* __restrict__ wt1,
                                  __bf16* __restrict__ wt2) {
    int idx = blockIdx.x * blockDim.x + threadIdx.x;
    if (idx >= 2 * 2 * 4 * 64 * 64) return;
    int k = idx & 63;
    int n = (idx >> 6) & 63;
    int bq = (idx >> 12) & 3;
    int comp = (idx >> 14) & 1;
    int layer = idx >> 15;
    const float* src = layer ? cw2 : cw1;
    __bf16* dst = layer ? wt2 : wt1;
    float v = (n < 48 && k < 48)
                  ? src[((comp * 4 + bq) * 48 + k) * 48 + n]
                  : 0.0f;
    dst[((comp * 4 + bq) * 64 + n) * 64 + k] = (__bf16)v;
}

// ---------------------------------------------------------------------------
// LayerNorm (one thread per row of 192)
// ---------------------------------------------------------------------------
__global__ void ln_kernel(const float* __restrict__ x,
                          const float* __restrict__ w,
                          const float* __restrict__ b,
                          float* __restrict__ outF,
                          __bf16* __restrict__ outB) {
    int m = blockIdx.x * blockDim.x + threadIdx.x;
    if (m >= MROWS) return;
    const float* row = x + (size_t)m * CDIM;
    float s = 0.0f, s2 = 0.0f;
    for (int c = 0; c < CDIM; ++c) { float v = row[c]; s += v; s2 += v * v; }
    float mu = s * (1.0f / CDIM);
    float var = s2 * (1.0f / CDIM) - mu * mu;
    float inv = rsqrtf(var + 1e-5f);
    for (int c = 0; c < CDIM; ++c) {
        float o = (row[c] - mu) * inv * w[c] + b[c];
        if (outF) outF[(size_t)m * CDIM + c] = o;
        if (outB) outB[(size_t)m * CDIM + c] = (__bf16)o;
    }
}

// ---------------------------------------------------------------------------
// causal depthwise conv1d (D_CONV=4) + SiLU.  xz: [4096][768], xm = cols 0:384
// ---------------------------------------------------------------------------
__global__ void conv_silu_kernel(const float* __restrict__ xz,
                                 const float* __restrict__ cw,
                                 const float* __restrict__ cb,
                                 float* __restrict__ xcF,
                                 __bf16* __restrict__ xcB) {
    int idx = blockIdx.x * blockDim.x + threadIdx.x;
    if (idx >= MROWS * DINNER) return;
    int d = idx % DINNER, m = idx / DINNER;
    int l = m & (LSEQ - 1), bq = m >> 10;
    float s = cb[d];
#pragma unroll
    for (int t = 0; t < 4; ++t) {
        int ls = l - 3 + t;
        if (ls >= 0)
            s += cw[d * 4 + t] * xz[(size_t)(bq * LSEQ + ls) * 768 + d];
    }
    float sil = s / (1.0f + __expf(-s));
    xcF[idx] = sil;
    xcB[idx] = (__bf16)sil;
}

// ---------------------------------------------------------------------------
// delta = softplus(dbc[:, :12] @ dt_proj_w^T + dt_proj_b)
// ---------------------------------------------------------------------------
__global__ void dt_softplus_kernel(const float* __restrict__ dbc,  // [4096][144]
                                   const float* __restrict__ dtw,  // [384][12]
                                   const float* __restrict__ dtb,
                                   float* __restrict__ delta) {
    int idx = blockIdx.x * blockDim.x + threadIdx.x;
    if (idx >= MROWS * DINNER) return;
    int d = idx % DINNER, m = idx / DINNER;
    float s = dtb[d];
#pragma unroll
    for (int r = 0; r < DTRANK; ++r)
        s += dbc[(size_t)m * 144 + r] * dtw[d * DTRANK + r];
    delta[idx] = (s > 20.0f) ? s : log1pf(__expf(s));
}

// ---------------------------------------------------------------------------
// selective scan: 1 block per batch, 1 thread per d-channel (h[64] in VGPRs)
// fused epilogue: y_out = (y + x*D) * silu(z)  -> bf16 for out_proj GEMM
// ---------------------------------------------------------------------------
__global__ void __launch_bounds__(DINNER)
scan_kernel(const float* __restrict__ dbc,    // [4096][144] (dt|B|C)
            const float* __restrict__ delta,  // [4096][384]
            const float* __restrict__ xc,     // [4096][384]
            const float* __restrict__ xz,     // [4096][768] (z = cols 384:)
            const float* __restrict__ A_log,  // [384][64]
            const float* __restrict__ Dw,     // [384]
            __bf16* __restrict__ ybf) {       // [4096][384]
    const int bq = blockIdx.x;
    const int d  = threadIdx.x;
    __shared__ float sB[DSTATE];
    __shared__ float sC[DSTATE];
    float h[DSTATE], A[DSTATE];
#pragma unroll
    for (int n = 0; n < DSTATE; ++n) {
        A[n] = -__expf(A_log[d * DSTATE + n]);
        h[n] = 0.0f;
    }
    const float Dd = Dw[d];
    for (int l = 0; l < LSEQ; ++l) {
        const int m = bq * LSEQ + l;
        if (d < 2 * DSTATE) {
            float v = dbc[(size_t)m * 144 + DTRANK + d];
            if (d < DSTATE) sB[d] = v; else sC[d - DSTATE] = v;
        }
        __syncthreads();
        const float dt = delta[(size_t)m * DINNER + d];
        const float xv = xc[(size_t)m * DINNER + d];
        const float du = dt * xv;
        float y = 0.0f;
#pragma unroll
        for (int n = 0; n < DSTATE; ++n) {
            float dA = __expf(dt * A[n]);
            h[n] = h[n] * dA + du * sB[n];
            y += h[n] * sC[n];
        }
        const float z = xz[(size_t)m * 768 + DINNER + d];
        const float sil = z / (1.0f + __expf(-z));
        ybf[(size_t)m * DINNER + d] = (__bf16)((y + xv * Dd) * sil);
        __syncthreads();
    }
}

// ---------------------------------------------------------------------------
// EinFFT stage 1: ortho size-4 FFT along block axis (real input)
// ---------------------------------------------------------------------------
__global__ void fft4_kernel(const float* __restrict__ xin,  // [4096][192]
                            float* __restrict__ Xr,
                            float* __restrict__ Xi) {
    int idx = blockIdx.x * blockDim.x + threadIdx.x;
    if (idx >= MROWS * 48) return;
    int d = idx % 48, m = idx / 48;
    const float* base = xin + (size_t)m * CDIM;
    float a0 = base[d], a1 = base[48 + d], a2 = base[96 + d], a3 = base[144 + d];
    float* rr = Xr + (size_t)m * CDIM;
    float* ii = Xi + (size_t)m * CDIM;
    rr[d]        = 0.5f * (a0 + a1 + a2 + a3); ii[d]        = 0.0f;
    rr[48 + d]   = 0.5f * (a0 - a2);           ii[48 + d]   = 0.5f * (a3 - a1);
    rr[96 + d]   = 0.5f * (a0 - a1 + a2 - a3); ii[96 + d]   = 0.0f;
    rr[144 + d]  = 0.5f * (a0 - a2);           ii[144 + d]  = 0.5f * (a1 - a3);
}

// ---------------------------------------------------------------------------
// 1024-point Stockham radix-2 FFT in LDS, one column per block.
// sign=-1 forward, +1 inverse; scale applied on store (ortho: 1/32).
// ---------------------------------------------------------------------------
__global__ void __launch_bounds__(512)
fft1024_kernel(float* __restrict__ re, float* __restrict__ im,
               float sign, float scale) {
    __shared__ float xr[1024], xi[1024], yr[1024], yi[1024];
    const int col = blockIdx.x % CDIM;
    const int Bq  = blockIdx.x / CDIM;
    const int t   = threadIdx.x;
    const size_t base = (size_t)Bq * LSEQ * CDIM + col;
    for (int e = t; e < 1024; e += 512) {
        xr[e] = re[base + (size_t)e * CDIM];
        xi[e] = im[base + (size_t)e * CDIM];
    }
    __syncthreads();
    float *sr = xr, *si = xi, *dr = yr, *di = yi;
    int l = 512, mm = 1;
#pragma unroll 1
    for (int st = 0; st < 10; ++st) {
        const int i = t / mm, j = t % mm;
        const float ang = sign * 3.14159265358979323846f * (float)i / (float)l;
        float sn, cs;
        __sincosf(ang, &sn, &cs);
        const int ia = i * mm + j;
        const int ib = ia + l * mm;
        const float ar = sr[ia], ai = si[ia];
        const float br = sr[ib], bi = si[ib];
        dr[2 * i * mm + j] = ar + br;
        di[2 * i * mm + j] = ai + bi;
        const float ur = ar - br, ui = ai - bi;
        dr[(2 * i + 1) * mm + j] = ur * cs - ui * sn;
        di[(2 * i + 1) * mm + j] = ur * sn + ui * cs;
        __syncthreads();
        float* tp;
        tp = sr; sr = dr; dr = tp;
        tp = si; si = di; di = tp;
        l >>= 1; mm <<= 1;
    }
    for (int e = t; e < 1024; e += 512) {
        re[base + (size_t)e * CDIM] = sr[e] * scale;
        im[base + (size_t)e * CDIM] = si[e] * scale;
    }
}

// ---------------------------------------------------------------------------
// Convert FFT output [4096][192] (blk*48+d) -> bf16 padded [4096][4*64]
// ---------------------------------------------------------------------------
__global__ void pad_bf16_kernel(const float* __restrict__ Xr,
                                const float* __restrict__ Xi,
                                __bf16* __restrict__ XrP,
                                __bf16* __restrict__ XiP) {
    int idx = blockIdx.x * blockDim.x + threadIdx.x;
    if (idx >= MROWS * 256) return;
    int cc = idx & 255, m = idx >> 8;
    int blk = cc >> 6, k = cc & 63;
    float vr = 0.0f, vi = 0.0f;
    if (k < 48) {
        vr = Xr[(size_t)m * CDIM + blk * 48 + k];
        vi = Xi[(size_t)m * CDIM + blk * 48 + k];
    }
    XrP[idx] = (__bf16)vr;
    XiP[idx] = (__bf16)vi;
}

// ---------------------------------------------------------------------------
// inverse size-4 FFT along block axis, real part, + residual -> final output
// ---------------------------------------------------------------------------
__global__ void final_kernel(const float* __restrict__ Zr,  // [4096][192]
                             const float* __restrict__ Zi,
                             const float* __restrict__ x1,  // [4096][192]
                             float* __restrict__ out) {
    int idx = blockIdx.x * blockDim.x + threadIdx.x;
    if (idx >= MROWS * 48) return;
    int d = idx % 48, m = idx / 48;
    const float* zr = Zr + (size_t)m * CDIM;
    const float* zi = Zi + (size_t)m * CDIM;
    float r0 = zr[d], r1 = zr[48 + d], r2 = zr[96 + d], r3 = zr[144 + d];
    float i1 = zi[48 + d], i3 = zi[144 + d];
    float y0 = 0.5f * (r0 + r1 + r2 + r3);
    float y1 = 0.5f * (r0 - i1 - r2 + i3);
    float y2 = 0.5f * (r0 - r1 + r2 - r3);
    float y3 = 0.5f * (r0 + i1 - r2 - i3);
    const size_t b = (size_t)m * CDIM;
    out[b + d]       = x1[b + d]       + y0;
    out[b + 48 + d]  = x1[b + 48 + d]  + y1;
    out[b + 96 + d]  = x1[b + 96 + d]  + y2;
    out[b + 144 + d] = x1[b + 144 + d] + y3;
}

// ---------------------------------------------------------------------------
// host side
// ---------------------------------------------------------------------------
static inline int cdiv(int a, int b) { return (a + b - 1) / b; }

extern "C" void kernel_launch(void* const* d_in, const int* in_sizes, int n_in,
                              void* d_out, int out_size, void* d_ws, size_t ws_size,
                              hipStream_t stream) {
    (void)in_sizes; (void)n_in; (void)out_size; (void)ws_size;
    const float* x        = (const float*)d_in[0];
    const float* ln1_w    = (const float*)d_in[1];
    const float* ln1_b    = (const float*)d_in[2];
    const float* in_proj  = (const float*)d_in[3];
    const float* conv_w   = (const float*)d_in[4];
    const float* conv_b   = (const float*)d_in[5];
    const float* x_proj   = (const float*)d_in[6];
    const float* dt_w     = (const float*)d_in[7];
    const float* dt_b     = (const float*)d_in[8];
    const float* A_log    = (const float*)d_in[9];
    const float* Dw       = (const float*)d_in[10];
    const float* out_proj = (const float*)d_in[11];
    const float* ln2_w    = (const float*)d_in[12];
    const float* ln2_b    = (const float*)d_in[13];
    const float* cw1      = (const float*)d_in[14];
    const float* cw2      = (const float*)d_in[15];
    const float* cb1      = (const float*)d_in[16];
    const float* cb2      = (const float*)d_in[17];
    float* out = (float*)d_out;

    char* ws = (char*)d_ws;
    size_t off = 0;
    auto alloc = [&](size_t bytes) -> void* {
        void* p = ws + off;
        off = (off + bytes + 255) & ~(size_t)255;
        return p;
    };

    __bf16* xln1_bf   = (__bf16*)alloc((size_t)MROWS * CDIM * 2);
    __bf16* w_inproj  = (__bf16*)alloc((size_t)768 * 192 * 2);
    __bf16* w_xproj   = (__bf16*)alloc((size_t)192 * 384 * 2);
    __bf16* w_outproj = (__bf16*)alloc((size_t)192 * 384 * 2);
    __bf16* wt1       = (__bf16*)alloc((size_t)2 * 4 * 64 * 64 * 2);
    __bf16* wt2       = (__bf16*)alloc((size_t)2 * 4 * 64 * 64 * 2);
    float*  xz        = (float*) alloc((size_t)MROWS * 768 * 4);
    float*  xcF       = (float*) alloc((size_t)MROWS * DINNER * 4);
    __bf16* xcB       = (__bf16*)alloc((size_t)MROWS * DINNER * 2);
    float*  dbc       = (float*) alloc((size_t)MROWS * 144 * 4);
    float*  delta     = (float*) alloc((size_t)MROWS * DINNER * 4);
    __bf16* y_bf      = (__bf16*)alloc((size_t)MROWS * DINNER * 2);
    float*  x1        = (float*) alloc((size_t)MROWS * CDIM * 4);
    float*  xln2      = (float*) alloc((size_t)MROWS * CDIM * 4);
    float*  Xr        = (float*) alloc((size_t)MROWS * CDIM * 4);  // reused as Zr
    float*  Xi        = (float*) alloc((size_t)MROWS * CDIM * 4);  // reused as Zi
    __bf16* XrP       = (__bf16*)alloc((size_t)MROWS * 256 * 2);
    __bf16* XiP       = (__bf16*)alloc((size_t)MROWS * 256 * 2);
    __bf16* r1P       = (__bf16*)alloc((size_t)MROWS * 256 * 2);
    __bf16* i1P       = (__bf16*)alloc((size_t)MROWS * 256 * 2);

    // ---- weight conversions -------------------------------------------------
    conv_f32_to_bf16<<<cdiv(768 * 192, 256), 256, 0, stream>>>(in_proj, w_inproj, 768 * 192);
    conv_xproj_pad<<<cdiv(192 * 384, 256), 256, 0, stream>>>(x_proj, w_xproj);
    conv_f32_to_bf16<<<cdiv(192 * 384, 256), 256, 0, stream>>>(out_proj, w_outproj, 192 * 384);
    conv_cw_transpose<<<cdiv(65536, 256), 256, 0, stream>>>(cw1, cw2, wt1, wt2);

    // ---- Mamba branch -------------------------------------------------------
    ln_kernel<<<cdiv(MROWS, 256), 256, 0, stream>>>(x, ln1_w, ln1_b, nullptr, xln1_bf);

    // in_proj: [4096,192] x [768,192]^T -> xz [4096,768]  (32x64 tiles)
    gemm_bf16_wmma<2, false><<<(128 * 12) / 4, 128, 0, stream>>>(
        xln1_bf, 192, w_inproj, 192, nullptr, nullptr, 0.0f,
        nullptr, nullptr, 0, xz, 768, nullptr, 0,
        MROWS, 768, 768, 192, 0);

    conv_silu_kernel<<<cdiv(MROWS * DINNER, 256), 256, 0, stream>>>(
        xz, conv_w, conv_b, xcF, xcB);

    // x_proj: [4096,384] x [192pad,384]^T -> dbc [4096,144] (valid 140)
    gemm_bf16_wmma<2, false><<<(128 * 3) / 4, 128, 0, stream>>>(
        xcB, 384, w_xproj, 384, nullptr, nullptr, 0.0f,
        nullptr, nullptr, 0, dbc, 144, nullptr, 0,
        MROWS, 192, 140, 384, 0);

    dt_softplus_kernel<<<cdiv(MROWS * DINNER, 256), 256, 0, stream>>>(
        dbc, dt_w, dt_b, delta);

    scan_kernel<<<NBATCH, DINNER, 0, stream>>>(
        dbc, delta, xcF, xz, A_log, Dw, y_bf);

    // out_proj + residual: x1 = x + y @ out_proj^T
    gemm_bf16_wmma<2, false><<<(128 * 3) / 4, 128, 0, stream>>>(
        y_bf, 384, w_outproj, 384, nullptr, nullptr, 0.0f,
        nullptr, x, CDIM, x1, CDIM, nullptr, 0,
        MROWS, 192, 192, 384, 0);

    // ---- EinFFT branch ------------------------------------------------------
    ln_kernel<<<cdiv(MROWS, 256), 256, 0, stream>>>(x1, ln2_w, ln2_b, xln2, nullptr);

    fft4_kernel<<<cdiv(MROWS * 48, 256), 256, 0, stream>>>(xln2, Xr, Xi);
    fft1024_kernel<<<NBATCH * CDIM, 512, 0, stream>>>(Xr, Xi, -1.0f, 1.0f / 32.0f);
    pad_bf16_kernel<<<cdiv(MROWS * 256, 256), 256, 0, stream>>>(Xr, Xi, XrP, XiP);

    // layer 1 (complex matmul + bias + relu), per block b
    for (int b = 0; b < 4; ++b) {
        const __bf16* W1r = wt1 + (size_t)b * 4096;
        const __bf16* W1i = wt1 + (size_t)(4 + b) * 4096;
        // r1 = relu(Xr@W1r - Xi@W1i + cb1[0])
        gemm_bf16_wmma<2, true><<<128 / 4, 128, 0, stream>>>(
            XrP + b * 64, 256, W1r, 64, XiP + b * 64, W1i, -1.0f,
            cb1 + b * 48, nullptr, 0, nullptr, 0, r1P + b * 64, 256,
            MROWS, 64, 48, 64, 1);
        // i1 = relu(Xr@W1i + Xi@W1r + cb1[1])
        gemm_bf16_wmma<2, true><<<128 / 4, 128, 0, stream>>>(
            XrP + b * 64, 256, W1i, 64, XiP + b * 64, W1r, 1.0f,
            cb1 + 192 + b * 48, nullptr, 0, nullptr, 0, i1P + b * 64, 256,
            MROWS, 64, 48, 64, 1);
    }
    // layer 2 (complex matmul + bias + softshrink) -> Zr/Zi (reuse Xr/Xi)
    for (int b = 0; b < 4; ++b) {
        const __bf16* W2r = wt2 + (size_t)b * 4096;
        const __bf16* W2i = wt2 + (size_t)(4 + b) * 4096;
        gemm_bf16_wmma<2, true><<<128 / 4, 128, 0, stream>>>(
            r1P + b * 64, 256, W2r, 64, i1P + b * 64, W2i, -1.0f,
            cb2 + b * 48, nullptr, 0, Xr + b * 48, CDIM, nullptr, 0,
            MROWS, 64, 48, 64, 2);
        gemm_bf16_wmma<2, true><<<128 / 4, 128, 0, stream>>>(
            r1P + b * 64, 256, W2i, 64, i1P + b * 64, W2r, 1.0f,
            cb2 + 192 + b * 48, nullptr, 0, Xi + b * 48, CDIM, nullptr, 0,
            MROWS, 64, 48, 64, 2);
    }

    fft1024_kernel<<<NBATCH * CDIM, 512, 0, stream>>>(Xr, Xi, 1.0f, 1.0f / 32.0f);
    final_kernel<<<cdiv(MROWS * 48, 256), 256, 0, stream>>>(Xr, Xi, x1, out);
}
